// SimVQ_79070347919417
// MI455X (gfx1250) — compile-verified, matrix operand order checked by reading, not run on hardware
//
#include <hip/hip_runtime.h>
#include <hip/hip_bf16.h>

#define B_  8
#define N_  2048
#define D_  512
#define C_  4096
#define BN_ (B_ * N_)

typedef __attribute__((ext_vector_type(2)))  float        v2f;
typedef __attribute__((ext_vector_type(8)))  float        v8f;
typedef __bf16 bf16t;
typedef __attribute__((ext_vector_type(16))) bf16t        v16bf;
typedef __attribute__((ext_vector_type(4)))  unsigned int v4u;
typedef __attribute__((ext_vector_type(4)))  int          v4i;
typedef __attribute__((ext_vector_type(8)))  int          v8i;

union BfFrag { v4u q[2]; v16bf v; };

__device__ __forceinline__ unsigned short f2bf(float f) {
    unsigned int u = __float_as_uint(f);
    u += 0x7FFFu + ((u >> 16) & 1u);          // round-to-nearest-even
    return (unsigned short)(u >> 16);
}

// ---------------------------------------------------------------------------
// Kernel 1: implicit = codebook @ W^T  (FP32, V_WMMA_F32_16X16X4_F32)
// One wave per 16x16 output tile; K=512 in steps of 4.
// ---------------------------------------------------------------------------
__global__ __launch_bounds__(32)
void simvq_implicit_gemm(const float* __restrict__ cb, const float* __restrict__ W,
                         float* __restrict__ imp) {
    const int lane = threadIdx.x & 31;
    const int r = lane & 15, h = lane >> 4;
    const int m0 = blockIdx.x * 16;   // codebook rows (M)
    const int n0 = blockIdx.y * 16;   // output feature cols (N)
    const float* arow = cb + (size_t)(m0 + r) * D_;
    const float* brow = W  + (size_t)(n0 + r) * D_;   // B[k][j] = W[j][k]
    v8f acc = {};
    for (int k0 = 0; k0 < D_; k0 += 4) {
        v2f a = *(const v2f*)(arow + k0 + 2 * h);
        v2f b = *(const v2f*)(brow + k0 + 2 * h);
        acc = __builtin_amdgcn_wmma_f32_16x16x4_f32(false, a, false, b,
                                                    (short)0, acc, false, false);
    }
    float* orow = imp + (size_t)(m0 + 8 * h) * D_ + n0 + r;
    #pragma unroll
    for (int j = 0; j < 8; ++j)
        orow[(size_t)j * D_] = acc[j];
}

// ---------------------------------------------------------------------------
// Kernel 2: BF16 copy of implicit (row-major [c][d]) + per-code squared norm
// ---------------------------------------------------------------------------
__global__ __launch_bounds__(256)
void simvq_prep_codes(const float* __restrict__ imp, unsigned short* __restrict__ impbf,
                      float* __restrict__ qnorm) {
    __shared__ float red[256];
    const int row = blockIdx.x, tid = threadIdx.x;
    const float* src = imp + (size_t)row * D_;
    unsigned short* dst = impbf + (size_t)row * D_;
    float s = 0.f;
    for (int i = tid; i < D_; i += 256) {
        float v = src[i];
        dst[i] = f2bf(v);
        s += v * v;
    }
    red[tid] = s; __syncthreads();
    for (int off = 128; off > 0; off >>= 1) {
        if (tid < off) red[tid] += red[tid + off];
        __syncthreads();
    }
    if (tid == 0) qnorm[row] = red[0];
}

// ---------------------------------------------------------------------------
// Kernel 2b: x -> bf16 (row-major [BN][D]) so the TDM can DMA tiles directly
// ---------------------------------------------------------------------------
__global__ __launch_bounds__(256)
void simvq_x_to_bf16(const float* __restrict__ x, unsigned short* __restrict__ xbf) {
    const size_t i0 = ((size_t)blockIdx.x * 256 + threadIdx.x) * 4;
    #pragma unroll
    for (int j = 0; j < 4; ++j)
        xbf[i0 + j] = f2bf(x[i0 + j]);
}

// ---------------------------------------------------------------------------
// Kernel 3: argmin over codes via BF16 WMMA (V_WMMA_F32_16X16X32_BF16)
// Block = 256 threads (8 waves) owns 32 rows of x, staged into LDS by the
// Tensor Data Mover. Each wave sweeps 4096 codes in 128-col tiles, 16 cols
// per wave; each B fragment feeds TWO wmma (row tiles 0-15 and 16-31).
// rank value = qnorm[n] - 2 * <x_m, code_n>   (||x||^2 constant per row)
// ---------------------------------------------------------------------------
__global__ __launch_bounds__(256)
void simvq_argmin(const unsigned short* __restrict__ xbf,
                  const unsigned short* __restrict__ impbf,
                  const float* __restrict__ qnorm, int* __restrict__ idxOut,
                  float* __restrict__ idxOutF) {
    __shared__ unsigned short Abf[32 * D_];   // 32 KB, LDS offset 0 (first alloc)
    __shared__ float wD[8][32];
    __shared__ int   wI[8][32];
    const int tid  = threadIdx.x;
    const int lane = tid & 31, wave = tid >> 5;
    const int m0 = blockIdx.x * 32;

    // --- TDM DMA: 32x512 bf16 tile of x -> LDS ------------------------------
    if (wave == 0) {
        unsigned long long ga = (unsigned long long)(const void*)(xbf + (size_t)m0 * D_);
        v4u g0; v8i g1; v4i g2 = {}, g3 = {};
        g0[0] = 1u;                                             // count=1
        g0[1] = 0u;                                             // lds_addr = 0 (Abf)
        g0[2] = (unsigned)(ga & 0xFFFFFFFFull);                 // global_addr lo
        g0[3] = (unsigned)((ga >> 32) & 0x1FFFFFFull) | (2u << 30); // hi | type=2
        g1[0] = 0x00010000;                 // workgroup_mask=0, data_size=1 (2B)
        g1[1] = (int)(512u << 16);          // tensor_dim0 = 512 (bits 79:48 lo16)
        g1[2] = (int)((unsigned)BN_ << 16); // tensor_dim1 = 16384 (bits 111:80 lo16)
        g1[3] = (int)(512u << 16);          // tile_dim0 = 512 (bits 127:112)
        g1[4] = 32;                         // tile_dim1 = 32, tile_dim2 = 0
        g1[5] = 512;                        // tensor_dim0_stride lo32
        g1[6] = 0; g1[7] = 0;
#if defined(__clang_major__) && (__clang_major__ >= 23)
        v8i g4 = {};
        __builtin_amdgcn_tensor_load_to_lds(g0, g1, g2, g3, g4, 0);
#else
        __builtin_amdgcn_tensor_load_to_lds(g0, g1, g2, g3, 0);
#endif
        __builtin_amdgcn_s_wait_tensorcnt(0);
    }
    __syncthreads();

    const int r = lane & 15, h = lane >> 4;
    const unsigned short* A0 = Abf + (size_t)r * D_;
    const unsigned short* A1 = Abf + (size_t)(16 + r) * D_;

    float bd0[8], bd1[8]; int bi0[8], bi1[8];
    #pragma unroll
    for (int j = 0; j < 8; ++j) {
        bd0[j] = 3.4e38f; bi0[j] = 0;
        bd1[j] = 3.4e38f; bi1[j] = 0;
    }

    for (int t = 0; t < C_ / 128; ++t) {
        const int ncol = t * 128 + wave * 16 + r;
        const unsigned short* brow = impbf + (size_t)ncol * D_;
        const float qn = qnorm[ncol];
        if (t + 1 < C_ / 128)
            __builtin_prefetch(brow + 128 * D_, 0, 1);   // global_prefetch_b8
        v8f acc0 = {}, acc1 = {};
        for (int k0 = 0; k0 < D_; k0 += 32) {
            // 16-bit 16x32 fragment = two contiguous 16B spans per lane:
            //   u[0..3] @ k0 + h*8 ; u[4..7] @ k0 + 16 + h*8
            BfFrag a0, a1, b;
            const int o0 = k0 + h * 8, o1 = k0 + 16 + h * 8;
            a0.q[0] = *(const v4u*)(A0 + o0);  a0.q[1] = *(const v4u*)(A0 + o1);
            a1.q[0] = *(const v4u*)(A1 + o0);  a1.q[1] = *(const v4u*)(A1 + o1);
            b.q[0]  = *(const v4u*)(brow + o0); b.q[1] = *(const v4u*)(brow + o1);
            acc0 = __builtin_amdgcn_wmma_f32_16x16x32_bf16(false, a0.v, false, b.v,
                                                           (short)0, acc0, false, false);
            acc1 = __builtin_amdgcn_wmma_f32_16x16x32_bf16(false, a1.v, false, b.v,
                                                           (short)0, acc1, false, false);
        }
        #pragma unroll
        for (int j = 0; j < 8; ++j) {
            float v0 = qn - 2.0f * acc0[j];
            float v1 = qn - 2.0f * acc1[j];
            if (v0 < bd0[j]) { bd0[j] = v0; bi0[j] = ncol; }
            if (v1 < bd1[j]) { bd1[j] = v1; bi1[j] = ncol; }
        }
    }

    // reduce across the 16 lanes holding each row (xor<16 stays in half)
    #pragma unroll
    for (int tt = 0; tt < 2; ++tt) {
        #pragma unroll
        for (int j = 0; j < 8; ++j) {
            float d = tt ? bd1[j] : bd0[j];
            int  ix = tt ? bi1[j] : bi0[j];
            #pragma unroll
            for (int off = 8; off >= 1; off >>= 1) {
                float od = __shfl_xor(d, off, 32);
                int   oi = __shfl_xor(ix, off, 32);
                if (od < d || (od == d && oi < ix)) { d = od; ix = oi; }
            }
            if (r == 0) {
                const int rowLocal = tt * 16 + j + 8 * h;
                wD[wave][rowLocal] = d; wI[wave][rowLocal] = ix;
            }
        }
    }
    __syncthreads();
    if (tid < 32) {
        float d = wD[0][tid]; int ix = wI[0][tid];
        for (int wv = 1; wv < 8; ++wv) {
            float od = wD[wv][tid]; int oi = wI[wv][tid];
            if (od < d || (od == d && oi < ix)) { d = od; ix = oi; }
        }
        idxOut[m0 + tid]  = ix;
        idxOutF[m0 + tid] = (float)ix;
    }
}

// ---------------------------------------------------------------------------
// Kernel 4: rotation trick + commit loss. One block (256 thr) per row.
// ---------------------------------------------------------------------------
__global__ __launch_bounds__(256)
void simvq_rotate(const float* __restrict__ x, const float* __restrict__ imp,
                  const int* __restrict__ idx, float* __restrict__ outQ,
                  float* __restrict__ outLoss) {
    __shared__ float s0[256], s1[256], s2[256], s3[256];
    __shared__ float sc[6];
    const int row = blockIdx.x, tid = threadIdx.x;
    const float* xr = x   + (size_t)row * D_;
    const float* qr = imp + (size_t)idx[row] * D_;
    float x0 = xr[tid], x1 = xr[tid + 256];
    float q0 = qr[tid], q1 = qr[tid + 256];
    float f0 = x0 - q0 + 1e-6f, f1 = x1 - q1 + 1e-6f;
    s0[tid] = x0 * x0 + x1 * x1;
    s1[tid] = q0 * q0 + q1 * q1;
    s2[tid] = x0 * q0 + x1 * q1;
    s3[tid] = f0 * f0 + f1 * f1;
    __syncthreads();
    for (int off = 128; off > 0; off >>= 1) {
        if (tid < off) {
            s0[tid] += s0[tid + off]; s1[tid] += s1[tid + off];
            s2[tid] += s2[tid + off]; s3[tid] += s3[tid + off];
        }
        __syncthreads();
    }
    if (tid == 0) {
        float sxx = s0[0], sqq = s1[0], sxq = s2[0];
        float nx = sqrtf(sxx), nq = sqrtf(sqq);
        float cx = fmaxf(nx, 1e-6f), cq = fmaxf(nq, 1e-6f);
        float nuq2 = sxx / (cx * cx) + 2.f * sxq / (cx * cq) + sqq / (cq * cq);
        float cw = fmaxf(sqrtf(fmaxf(nuq2, 0.f)), 1e-12f);
        sc[0] = 1.f / cx;
        sc[1] = 1.f / cq;
        sc[2] = 1.f / cw;
        sc[3] = (sxx / cx + sxq / cq) / cw;   // e_dot_w
        sc[4] = sxx / cx;                     // e_dot_u
        sc[5] = nq / cx;                      // output scale
        atomicAdd(outLoss, s3[0] * (1.0f / (float)BN_));
    }
    __syncthreads();
    const float icx = sc[0], icq = sc[1], icw = sc[2];
    const float edw = sc[3], edu = sc[4], scl = sc[5];
    float* o = outQ + (size_t)row * D_;
    float w0 = (x0 * icx + q0 * icq) * icw;
    float w1 = (x1 * icx + q1 * icq) * icw;
    o[tid]       = (x0 - 2.f * edw * w0 + 2.f * edu * (q0 * icq)) * scl;
    o[tid + 256] = (x1 - 2.f * edw * w1 + 2.f * edu * (q1 * icq)) * scl;
}

// ---------------------------------------------------------------------------
extern "C" void kernel_launch(void* const* d_in, const int* in_sizes, int n_in,
                              void* d_out, int out_size, void* d_ws, size_t ws_size,
                              hipStream_t stream) {
    const float* x  = (const float*)d_in[0];   // [8,2048,512]
    const float* cb = (const float*)d_in[1];   // [4096,512]
    const float* W  = (const float*)d_in[2];   // [512,512]
    float* out = (float*)d_out;

    char* ws = (char*)d_ws;
    float*          imp   = (float*)ws;                                  // 8 MB
    unsigned short* impbf = (unsigned short*)(ws + (size_t)C_ * D_ * 4); // 4 MB
    float*          qnorm = (float*)(ws + (size_t)C_ * D_ * 6);          // 16 KB
    int*            idx   = (int*)(ws + (size_t)C_ * D_ * 6 + (size_t)C_ * 4);
    unsigned short* xbf   = (unsigned short*)(ws + (size_t)C_ * D_ * 6
                                                 + (size_t)C_ * 4 + (size_t)BN_ * 4); // 16 MB

    float* outQ    = out;
    float* outIdxF = out + (size_t)BN_ * D_;
    float* outLoss = out + (size_t)BN_ * D_ + BN_;

    simvq_implicit_gemm<<<dim3(C_ / 16, D_ / 16), 32, 0, stream>>>(cb, W, imp);
    simvq_prep_codes<<<C_, 256, 0, stream>>>(imp, impbf, qnorm);
    simvq_x_to_bf16<<<(BN_ * D_) / 1024, 256, 0, stream>>>(x, xbf);
    simvq_argmin<<<BN_ / 32, 256, 0, stream>>>(xbf, impbf, qnorm, idx, outIdxF);
    hipMemsetAsync(outLoss, 0, sizeof(float), stream);
    simvq_rotate<<<BN_, 256, 0, stream>>>(x, imp, idx, outQ, outLoss);
}
